// SetConvGridEncoder_54683523612695
// MI455X (gfx1250) — compile-verified, matrix operand order checked by reading, not run on hardware
//
#include <hip/hip_runtime.h>
#include <stdint.h>

typedef __attribute__((ext_vector_type(16))) _Float16 v16h;
typedef __attribute__((ext_vector_type(8)))  _Float16 v8h;
typedef __attribute__((ext_vector_type(4)))  _Float16 v4h;
typedef __attribute__((ext_vector_type(8)))  float    v8f;

// Problem constants (from reference setup_inputs)
#define BATCH 4
#define NPTS  8192            // off-grid points (K dimension of GEMM)
#define DXDIM 2
#define DZ    128             // output feature dim
#define GM    2304            // 48*48 grid points (M dimension)
#define MT    32              // M tile per workgroup (2 wmma row-tiles)
#define KCO   64              // K chunk per LDS stage (= 2 WMMA k-steps of 32)
#define NCH   (NPTS / KCO)    // 128 outer iterations
#define MTILES (GM / MT)      // 72
#define WS_BYTES ((size_t)BATCH * DZ * NPTS * sizeof(_Float16))   // 8 MB

// CDNA5 async DMA: global -> LDS without VGPR transit, tracked by ASYNCcnt.
#define ASYNC_LDS_B128(ldsoff, gaddr)                                        \
    asm volatile("global_load_async_to_lds_b128 %0, %1, off"                 \
                 :: "v"(ldsoff), "v"(gaddr) : "memory")
#define WAIT_ASYNCCNT0() asm volatile("s_wait_asynccnt 0x0" ::: "memory")

// ---------------------------------------------------------------------------
// Pre-pass: zc_off [B][N][DZ] f32  ->  ws [B][DZ][N] f16  (convert + transpose)
// One-time ~24 MB of traffic; amortized over the 72 M-tile blocks per batch.
// ---------------------------------------------------------------------------
__global__ __launch_bounds__(256)
void zc_transpose_f16_kernel(const float* __restrict__ zc_off,
                             _Float16* __restrict__ ws)
{
    __shared__ _Float16 sT[DZ * 128];         // [d][n] tile, 32 KB
    const int bb = blockIdx.x >> 6;           // 64 n-tiles of 128 per batch
    const int n0 = (blockIdx.x & 63) * 128;
    const int tid = threadIdx.x;

    const int d4  = tid & 31;                 // group of 4 d columns
    const int nr4 = tid >> 5;                 // group of 4 n rows
    #pragma unroll
    for (int s = 0; s < 4; ++s) {
        const int nl = s * 32 + nr4 * 4;
        const float4* src = (const float4*)(zc_off +
                            ((size_t)(bb * NPTS + n0 + nl)) * DZ) + d4;
        const float4 r0 = src[0 * (DZ / 4)];
        const float4 r1 = src[1 * (DZ / 4)];
        const float4 r2 = src[2 * (DZ / 4)];
        const float4 r3 = src[3 * (DZ / 4)];
        _Float16* dst = &sT[(d4 * 4) * 128 + nl];
        v4h t0 = {(_Float16)r0.x, (_Float16)r1.x, (_Float16)r2.x, (_Float16)r3.x};
        v4h t1 = {(_Float16)r0.y, (_Float16)r1.y, (_Float16)r2.y, (_Float16)r3.y};
        v4h t2 = {(_Float16)r0.z, (_Float16)r1.z, (_Float16)r2.z, (_Float16)r3.z};
        v4h t3 = {(_Float16)r0.w, (_Float16)r1.w, (_Float16)r2.w, (_Float16)r3.w};
        *(v4h*)(dst + 0 * 128) = t0;
        *(v4h*)(dst + 1 * 128) = t1;
        *(v4h*)(dst + 2 * 128) = t2;
        *(v4h*)(dst + 3 * 128) = t3;
    }
    __syncthreads();
    #pragma unroll
    for (int e = 0; e < 4; ++e) {             // 1024 x 16B segments out
        const int flat = tid + e * 256;
        const int d = flat >> 3, seg = flat & 7;
        const v8h v = *(const v8h*)&sT[d * 128 + seg * 8];
        *(v8h*)(ws + ((size_t)bb * DZ + d) * NPTS + n0 + seg * 8) = v;
    }
}

// ---------------------------------------------------------------------------
// Main kernel: 32x128 output slab per block; 8 waves x (2 row-tiles) x 16 dz.
// USE_WS=true: B tiles DMA'd from pre-converted f16 workspace (no VALU).
// ---------------------------------------------------------------------------
template<bool USE_WS>
__global__ __launch_bounds__(256)
void setconv_wmma_kernel(const float* __restrict__ xc_off,   // [B, N, 2]
                         const float* __restrict__ xc_on,    // [B, GM, 2]
                         const float* __restrict__ zc_off,   // [B, N, DZ]
                         const _Float16* __restrict__ zc16,  // [B, DZ, N] (ws)
                         const float* __restrict__ zc_on,    // [B, GM, DZ]
                         const float* __restrict__ ls_param, // [2]
                         const int*   __restrict__ ignore_flag,
                         float* __restrict__ out_z)          // [B, GM, DZ]
{
    __shared__ _Float16 sA[2][MT * KCO];   // weight tile [m][k]         (8 KB)
    __shared__ _Float16 sB[2][DZ * KCO];   // zc chunk [d][k] f16        (32 KB)
    __shared__ float2   sX2[2][KCO];       // raw off-grid coords (async, 1 KB)
    __shared__ float    sX1[MT * DXDIM];   // scaled grid coords of this M tile

    const int tid  = threadIdx.x;
    const int lane = tid & 31;
    const int wave = tid >> 5;              // 0..7 -> dz tile of 16
    const int bb    = blockIdx.x / MTILES;
    const int mbase = (blockIdx.x % MTILES) * MT;

    // lengthscale = 1e-5 + softplus(param); keep reciprocals (loop-invariant)
    const float rls0 = 1.0f / (1e-5f + __logf(1.0f + __expf(ls_param[0])));
    const float rls1 = 1.0f / (1e-5f + __logf(1.0f + __expf(ls_param[1])));
    const int   ign  = ignore_flag[0];

    if (tid < MT * DXDIM) {
        float v = xc_on[(size_t)(bb * GM + mbase) * DXDIM + tid];
        sX1[tid] = v * ((tid & 1) ? rls1 : rls0);
    }

    const uintptr_t x2_gbase = (uintptr_t)(xc_off + (size_t)bb * NPTS * DXDIM);
    const uintptr_t b_gbase  = (uintptr_t)(zc16 + (size_t)bb * DZ * NPTS);

    auto asyncX2 = [&](int xbuf, int chunk) {   // 512B raw f32 coords
        const uint32_t lbase = (uint32_t)(uintptr_t)&sX2[xbuf][0];
        const uint64_t gb    = x2_gbase + (size_t)chunk * KCO * sizeof(float2);
        ASYNC_LDS_B128(lbase + lane * 16, gb + (uint64_t)(lane * 16));
    };
    auto asyncB = [&](int buf, int nbase) {     // 16 KB f16 B tile, 4 b128/thread
        const uint32_t lb = (uint32_t)(uintptr_t)&sB[buf][0];
        const uint64_t gb = b_gbase + (size_t)nbase * sizeof(_Float16);
        #pragma unroll
        for (int e = 0; e < 4; ++e) {
            const int flat = tid + e * 256;     // 1024 x 16B segments
            const int d = flat >> 3, seg = flat & 7;
            ASYNC_LDS_B128(lb + (uint32_t)(d * (KCO * 2) + seg * 16),
                           gb + (size_t)d * (NPTS * 2) + (size_t)(seg * 16));
        }
    };
    auto stageA = [&](int buf, int xbuf) {      // 32x64 exp-weight tile
        #pragma unroll
        for (int e = 0; e < 8; ++e) {
            const int idx = tid + e * 256;      // idx = m*64 + k
            const int m = idx >> 6;
            const int k = idx & 63;
            const float2 x2 = sX2[xbuf][k];
            const float d0 = sX1[2 * m]     - x2.x * rls0;
            const float d1 = sX1[2 * m + 1] - x2.y * rls1;
            sA[buf][m * KCO + k] = (_Float16)__expf(-0.5f * (d0 * d0 + d1 * d1));
        }
    };
    auto stageB = [&](int buf, int nbase) {     // fallback: convert in-kernel
        const int d4  = tid & 31;
        const int kr4 = tid >> 5;
        #pragma unroll
        for (int s = 0; s < 2; ++s) {
            const int k0 = s * 32 + kr4 * 4;
            const float4* src = (const float4*)(zc_off +
                                (size_t)(bb * NPTS + nbase + k0) * DZ) + d4;
            const float4 r0 = src[0 * (DZ / 4)];
            const float4 r1 = src[1 * (DZ / 4)];
            const float4 r2 = src[2 * (DZ / 4)];
            const float4 r3 = src[3 * (DZ / 4)];
            _Float16* dst = &sB[buf][(d4 * 4) * KCO + k0];
            v4h t0 = {(_Float16)r0.x, (_Float16)r1.x, (_Float16)r2.x, (_Float16)r3.x};
            v4h t1 = {(_Float16)r0.y, (_Float16)r1.y, (_Float16)r2.y, (_Float16)r3.y};
            v4h t2 = {(_Float16)r0.z, (_Float16)r1.z, (_Float16)r2.z, (_Float16)r3.z};
            v4h t3 = {(_Float16)r0.w, (_Float16)r1.w, (_Float16)r2.w, (_Float16)r3.w};
            *(v4h*)(dst + 0 * KCO) = t0;
            *(v4h*)(dst + 1 * KCO) = t1;
            *(v4h*)(dst + 2 * KCO) = t2;
            *(v4h*)(dst + 3 * KCO) = t3;
        }
    };

    // ---- Prologue ----
    if (wave == 0) { asyncX2(0, 0); asyncX2(1, 1); }
    if constexpr (USE_WS) asyncB(0, 0);
    WAIT_ASYNCCNT0();
    __syncthreads();              // x2 chunks (+B0) + sX1 visible to all waves
    stageA(0, 0);
    if constexpr (!USE_WS) stageB(0, 0);
    __syncthreads();

    v8f acc0 = {};                // rows mbase+0..15
    v8f acc1 = {};                // rows mbase+16..31
    // Fragment geometry (ISA 16-bit A/B layout, wave32):
    //   lane L: row/col = L%16, kb = 8*(L/16)
    //   halves 0..7 <- K kb..kb+7 ; halves 8..15 <- K kb+16..kb+23
    const int frag_rc = lane & 15;
    const int kb      = (lane >> 4) << 3;
    const int dzcol   = wave * 16 + frag_rc;

    for (int c = 0; c < NCH; ++c) {
        const int cur = c & 1;

        // Async x2 for chunk c+2 (consumed by stageA at iter c+1)
        if (wave == 0 && c + 2 < NCH) asyncX2(cur, c + 2);
        // Stage chunk c+1 into the other buffer (overlaps with WMMA below)
        if (c + 1 < NCH) {
            if constexpr (USE_WS) asyncB(cur ^ 1, (c + 1) * KCO);
            else                  stageB(cur ^ 1, (c + 1) * KCO);
            stageA(cur ^ 1, (c + 1) & 1);
        }

        // Consume current buffer: 2 k-steps x 2 row-tiles (B reused in regs)
        #pragma unroll
        for (int s = 0; s < 2; ++s) {
            const _Float16* ap0 = &sA[cur][frag_rc        * KCO + s * 32 + kb];
            const _Float16* ap1 = &sA[cur][(16 + frag_rc) * KCO + s * 32 + kb];
            const _Float16* bp  = &sB[cur][dzcol          * KCO + s * 32 + kb];
            const v8h b_lo  = *(const v8h*)bp;             // ds_load_b128
            const v8h b_hi  = *(const v8h*)(bp + 16);
            const v8h a0_lo = *(const v8h*)ap0;
            const v8h a0_hi = *(const v8h*)(ap0 + 16);
            const v8h a1_lo = *(const v8h*)ap1;
            const v8h a1_hi = *(const v8h*)(ap1 + 16);
            const v16h bm = __builtin_shufflevector(b_lo, b_hi,
                              0,1,2,3,4,5,6,7,8,9,10,11,12,13,14,15);
            const v16h a0 = __builtin_shufflevector(a0_lo, a0_hi,
                              0,1,2,3,4,5,6,7,8,9,10,11,12,13,14,15);
            const v16h a1 = __builtin_shufflevector(a1_lo, a1_hi,
                              0,1,2,3,4,5,6,7,8,9,10,11,12,13,14,15);
            acc0 = __builtin_amdgcn_wmma_f32_16x16x32_f16(
                       false, a0, false, bm, (short)0, acc0, false, false);
            acc1 = __builtin_amdgcn_wmma_f32_16x16x32_f16(
                       false, a1, false, bm, (short)0, acc1, false, false);
        }

        WAIT_ASYNCCNT0();     // async DMAs for next chunks have landed
        __syncthreads();      // publish stage (c+1), retire reads of buffer cur
    }

    // ---- Epilogue: C/D layout -> lane col = lane%16, rows r + 8*(lane/16) ----
    const int d = wave * 16 + (lane & 15);
    #pragma unroll
    for (int msub = 0; msub < 2; ++msub) {
        const v8f acc = msub ? acc1 : acc0;
        const int mrow0 = mbase + msub * 16 + ((lane >> 4) << 3);
        #pragma unroll
        for (int r = 0; r < 8; ++r) {
            const size_t idx = ((size_t)(bb * GM + mrow0 + r)) * DZ + d;
            float v = acc[r];
            if (!ign) v += zc_on[idx];
            out_z[idx] = v;
        }
    }
}

extern "C" void kernel_launch(void* const* d_in, const int* in_sizes, int n_in,
                              void* d_out, int out_size, void* d_ws, size_t ws_size,
                              hipStream_t stream) {
    const float* xc_off = (const float*)d_in[0];   // [4,8192,2]
    const float* xc_on  = (const float*)d_in[1];   // [4,48,48,2]
    const float* zc_off = (const float*)d_in[2];   // [4,8192,128]
    const float* zc_on  = (const float*)d_in[3];   // [4,48,48,128]
    const float* lsp    = (const float*)d_in[4];   // [2]
    const int*   ign    = (const int*)d_in[5];     // [1]
    float* out = (float*)d_out;
    _Float16* ws = (_Float16*)d_ws;

    // Output tuple: (xc_on_grid, z_grid) concatenated flat.
    const size_t xc_on_elems = (size_t)BATCH * GM * DXDIM;   // 18432
    hipMemcpyAsync(out, xc_on, xc_on_elems * sizeof(float),
                   hipMemcpyDeviceToDevice, stream);

    dim3 grid(BATCH * MTILES);   // 288 workgroups (32x128 output slab each)
    dim3 block(256);             // 8 wave32

    if (ws_size >= WS_BYTES) {
        // Pre-convert zc_off to f16 [b][d][n]; main kernel DMA-feeds B tiles.
        hipLaunchKernelGGL(zc_transpose_f16_kernel, dim3(BATCH * (NPTS / 128)),
                           block, 0, stream, zc_off, ws);
        hipLaunchKernelGGL((setconv_wmma_kernel<true>), grid, block, 0, stream,
                           xc_off, xc_on, zc_off, ws, zc_on, lsp, ign,
                           out + xc_on_elems);
    } else {
        hipLaunchKernelGGL((setconv_wmma_kernel<false>), grid, block, 0, stream,
                           xc_off, xc_on, zc_off, ws, zc_on, lsp, ign,
                           out + xc_on_elems);
    }
}
